// Edge_Encoder_51573967290599
// MI455X (gfx1250) — compile-verified, matrix operand order checked by reading, not run on hardware
//
#include <hip/hip_runtime.h>
#include <math.h>

// ---------------------------------------------------------------------------
// Problem constants (match reference)
// ---------------------------------------------------------------------------
#define NN    50000
#define EE    800000
#define EF    (EE + NN)          // edges + self loops
#define FEDGE 16
#define HC    128                // heads*hid
#define NHEAD 4
#define HID   32

typedef __attribute__((ext_vector_type(16))) _Float16 v16h;
typedef __attribute__((ext_vector_type(8)))  _Float16 v8h;
typedef __attribute__((ext_vector_type(8)))  float    v8f;

// ---------------------------------------------------------------------------
// Small utility kernels
// ---------------------------------------------------------------------------
__global__ void fill_f32(float* __restrict__ p, float v, int n) {
    int i = blockIdx.x * blockDim.x + threadIdx.x;
    if (i < n) p[i] = v;
}

__global__ void f32_to_f16(const float* __restrict__ in, _Float16* __restrict__ out, int n) {
    int i = blockIdx.x * blockDim.x + threadIdx.x;
    if (i < n) out[i] = (_Float16)in[i];
}

// ---------------------------------------------------------------------------
// Pack W[K x 128] (f32, row-major) into B-fragment order (f16) so the GEMM
// reads each lane's 16 halves as two contiguous b128 loads.
// Layout: Wp[(((wave*KSTEPS + ks)*32 + lane)*16) + j]
//   n = wave*16 + (lane&15); kb = (lane<16)?0:8
//   k = ks*32 + kb + (j<8 ? j : j+8)
// ---------------------------------------------------------------------------
__global__ void pack_w_f16(const float* __restrict__ W, _Float16* __restrict__ Wp, int K) {
    int i = blockIdx.x * blockDim.x + threadIdx.x;
    if (i >= K * HC) return;
    int j      = i & 15;
    int l      = (i >> 4) & 31;
    int kbt    = i >> 9;
    int ksteps = K >> 5;
    int wave   = kbt / ksteps;
    int ks     = kbt % ksteps;
    int n  = wave * 16 + (l & 15);
    int kk = ks * 32 + ((l < 16) ? 0 : 8) + ((j < 8) ? j : j + 8);
    Wp[i] = (_Float16)W[(size_t)kk * HC + n];
}

// ---------------------------------------------------------------------------
// Self-loop attr: per-target mean of incoming edge_attr
// ---------------------------------------------------------------------------
__global__ void edge_count_sum(const int* __restrict__ tgt, const float* __restrict__ ea,
                               float* __restrict__ cnt, float* __restrict__ asum) {
    int i = blockIdx.x * blockDim.x + threadIdx.x;
    if (i >= EE * FEDGE) return;
    int e = i >> 4;
    int k = i & 15;
    int t = tgt[e];
    atomicAdd(&asum[(size_t)t * FEDGE + k], ea[i]);
    if (k == 0) atomicAdd(&cnt[t], 1.0f);
}

__global__ void make_loop_attr(const float* __restrict__ cnt, const float* __restrict__ asum,
                               float* __restrict__ lattr) {
    int i = blockIdx.x * blockDim.x + threadIdx.x;
    if (i >= NN * FEDGE) return;
    int n = i >> 4;
    lattr[i] = asum[i] / fmaxf(cnt[n], 1.0f);
}

// ---------------------------------------------------------------------------
// WMMA GEMM: C[NN x 128] = A[NN x K](f16 row-major) * W(packed f16) + bias
// 3125 row-tiles = 5^5, so each wave owns MT=5 M-tiles (80 rows / block).
// block = 256 (8 waves = 8 column tiles); grid.x = NN/80 = 625.
// B fragment loaded once per K-step and reused by 5 independent WMMA chains.
// Fully unrolled via template<K> (K in {64,128}).
// ---------------------------------------------------------------------------
template <int K>
__global__ void gemm_wmma_f16(const _Float16* __restrict__ A, const _Float16* __restrict__ Wp,
                              const float* __restrict__ bias, float* __restrict__ C) {
    constexpr int KSTEPS = K / 32;
    constexpr int MT     = 5;
    const int blk  = blockIdx.x;
    const int wave = threadIdx.x >> 5;        // column tile 0..7
    const int lane = threadIdx.x & 31;
    const int n    = wave * 16 + (lane & 15);
    const int kb   = (lane < 16) ? 0 : 8;
    const int m0   = blk * (16 * MT) + (lane & 15);

    v8f acc[MT] = {};
#pragma unroll
    for (int ks = 0; ks < KSTEPS; ++ks) {
        const _Float16* bp = Wp + (((size_t)wave * KSTEPS + ks) * 32 + lane) * 16;
        v8h  blo = *(const v8h*)bp;
        v8h  bhi = *(const v8h*)(bp + 8);
        v16h b   = __builtin_shufflevector(blo, bhi, 0,1,2,3,4,5,6,7,8,9,10,11,12,13,14,15);
#pragma unroll
        for (int t = 0; t < MT; ++t) {
            const _Float16* ar = A + (size_t)(m0 + t * 16) * K + ks * 32 + kb;
            v8h  alo = *(const v8h*)ar;
            v8h  ahi = *(const v8h*)(ar + 16);
            v16h a   = __builtin_shufflevector(alo, ahi, 0,1,2,3,4,5,6,7,8,9,10,11,12,13,14,15);
            acc[t] = __builtin_amdgcn_wmma_f32_16x16x32_f16(
                /*neg_a=*/false, a, /*neg_b=*/false, b,
                /*c_mod=*/(short)0, acc[t], /*reuse_a=*/false, /*reuse_b=*/false);
        }
    }
    const float bv = bias[n];
    const int   rb = blk * (16 * MT) + ((lane >> 4) << 3);
#pragma unroll
    for (int t = 0; t < MT; ++t) {
#pragma unroll
        for (int i = 0; i < 8; ++i)
            C[(size_t)(rb + t * 16 + i) * HC + n] = acc[t][i] + bv;
    }
}

// ---------------------------------------------------------------------------
// Float atomic max via CAS (logits can be negative)
// ---------------------------------------------------------------------------
__device__ inline void atomicMaxF(float* addr, float v) {
    unsigned int* ua  = (unsigned int*)addr;
    unsigned int  old = *ua;
    while (__uint_as_float(old) < v) {
        unsigned int assumed = old;
        old = atomicCAS(ua, assumed, __float_as_uint(v));
        if (old == assumed) break;
    }
}

// ---------------------------------------------------------------------------
// Edge pass 1: logits + per-target max. One wave32 per edge; lane c owns
// channels {c, c+32, c+64, c+96} (one per head) -> coalesced row loads,
// butterfly reduction for the per-head dot with att.
// Fuses ee = edge_attr @ We (K=16) inline; self-loop rows use loop_attr.
// ---------------------------------------------------------------------------
__global__ void edge_logit(const int* __restrict__ src, const int* __restrict__ tgt,
                           const float* __restrict__ ea, const float* __restrict__ lattr,
                           const float* __restrict__ xl, const float* __restrict__ xr,
                           const float* __restrict__ We, const float* __restrict__ att,
                           float* __restrict__ logit, float* __restrict__ mx) {
    int e = blockIdx.x * (blockDim.x >> 5) + (threadIdx.x >> 5);
    if (e >= EF) return;
    int lane = threadIdx.x & 31;
    int s, t;
    const float4* er4;
    if (e < EE) { s = src[e]; t = tgt[e]; er4 = (const float4*)(ea + (size_t)e * FEDGE); }
    else        { s = t = e - EE;         er4 = (const float4*)(lattr + (size_t)s * FEDGE); }
    float er[FEDGE];
#pragma unroll
    for (int q = 0; q < 4; ++q) {
        float4 v = er4[q];
        er[q * 4 + 0] = v.x; er[q * 4 + 1] = v.y; er[q * 4 + 2] = v.z; er[q * 4 + 3] = v.w;
    }

    float m[NHEAD];
#pragma unroll
    for (int h = 0; h < NHEAD; ++h) {
        int   c = h * HID + lane;
        float v = 0.0f;
#pragma unroll
        for (int k = 0; k < FEDGE; ++k) v = fmaf(er[k], We[k * HC + c], v);
        v += xl[(size_t)s * HC + c] + xr[(size_t)t * HC + c];
        v = (v > 0.0f) ? v : 0.2f * v;           // leaky_relu
        m[h] = v * att[c];
    }
#pragma unroll
    for (int off = 16; off >= 1; off >>= 1) {
#pragma unroll
        for (int h = 0; h < NHEAD; ++h) m[h] += __shfl_xor(m[h], off, 32);
    }
    if (lane < NHEAD) {
        logit[(size_t)e * NHEAD + lane] = m[lane];
        atomicMaxF(&mx[(size_t)t * NHEAD + lane], m[lane]);
    }
}

// ---------------------------------------------------------------------------
// Edge pass 2: p = exp(logit - mx[tgt]); accumulate den and agg += xl[src]*p.
// Normalization by den happens once per node in the finalize kernel
// (alpha = p/den distributes over the segment sum).
// ---------------------------------------------------------------------------
__global__ void edge_aggregate(const int* __restrict__ src, const int* __restrict__ tgt,
                               const float* __restrict__ xl,
                               const float* __restrict__ logit, const float* __restrict__ mx,
                               float* __restrict__ den, float* __restrict__ agg) {
    int e = blockIdx.x * (blockDim.x >> 5) + (threadIdx.x >> 5);
    if (e >= EF) return;
    int lane = threadIdx.x & 31;
    int s, t;
    if (e < EE) { s = src[e]; t = tgt[e]; }
    else        { s = t = e - EE; }

    float4 lg = *(const float4*)(logit + (size_t)e * NHEAD);
    float4 mv = *(const float4*)(mx + (size_t)t * NHEAD);
    float p[NHEAD];
    p[0] = __expf(lg.x - mv.x);
    p[1] = __expf(lg.y - mv.y);
    p[2] = __expf(lg.z - mv.z);
    p[3] = __expf(lg.w - mv.w);

    if (lane < NHEAD) atomicAdd(&den[(size_t)t * NHEAD + lane], p[lane]);
#pragma unroll
    for (int h = 0; h < NHEAD; ++h) {
        int c = h * HID + lane;
        atomicAdd(&agg[(size_t)t * HC + c], xl[(size_t)s * HC + c] * p[h]);
    }
}

// ---------------------------------------------------------------------------
// Node finalize: h = relu(agg/den + bias)
// ---------------------------------------------------------------------------
__global__ void node_finalize(const float* __restrict__ agg, const float* __restrict__ den,
                              const float* __restrict__ bias, float* __restrict__ out) {
    int i = blockIdx.x * blockDim.x + threadIdx.x;
    if (i >= NN * HC) return;
    int n = i >> 7;
    int c = i & (HC - 1);
    int h = c >> 5;
    float v = agg[i] / den[(size_t)n * NHEAD + h] + bias[c];
    out[i] = v > 0.0f ? v : 0.0f;
}

// ---------------------------------------------------------------------------
// Host-side launch
// ---------------------------------------------------------------------------
static inline int cdiv(int a, int b) { return (a + b - 1) / b; }

extern "C" void kernel_launch(void* const* d_in, const int* in_sizes, int n_in,
                              void* d_out, int out_size, void* d_ws, size_t ws_size,
                              hipStream_t stream) {
    const float* x    = (const float*)d_in[0];
    const int*   ei   = (const int*)d_in[1];
    const float* ea   = (const float*)d_in[2];
    const float* w1l  = (const float*)d_in[3];
    const float* b1l  = (const float*)d_in[4];
    const float* w1r  = (const float*)d_in[5];
    const float* b1r  = (const float*)d_in[6];
    const float* w1e  = (const float*)d_in[7];
    const float* att1 = (const float*)d_in[8];
    const float* bias1= (const float*)d_in[9];
    const float* w2l  = (const float*)d_in[10];
    const float* b2l  = (const float*)d_in[11];
    const float* w2r  = (const float*)d_in[12];
    const float* b2r  = (const float*)d_in[13];
    const float* w2e  = (const float*)d_in[14];
    const float* att2 = (const float*)d_in[15];
    const float* bias2= (const float*)d_in[16];
    const float* hw   = (const float*)d_in[17];
    const float* hb   = (const float*)d_in[18];
    float* out = (float*)d_out;

    const int* src = ei;
    const int* tgt = ei + EE;

    // ---- workspace carve (all chunks 16B-aligned: sizes are multiples of 16 elems) ----
    char*  ws  = (char*)d_ws;
    size_t off = 0;
    auto carveF = [&](size_t nelem) { float* p = (float*)(ws + off); off += nelem * sizeof(float); return p; };
    float* cnt   = carveF(NN);                  // + asum contiguous (zeroed together)
    float* asum  = carveF((size_t)NN * FEDGE);
    float* lattr = carveF((size_t)NN * FEDGE);
    float* xl    = carveF((size_t)NN * HC);
    float* xr    = carveF((size_t)NN * HC);
    float* agg   = carveF((size_t)NN * HC);
    float* hbuf  = carveF((size_t)NN * HC);
    float* logit = carveF((size_t)EF * NHEAD);
    float* mx    = carveF((size_t)NN * NHEAD);
    float* den   = carveF((size_t)NN * NHEAD);
    _Float16* xh = (_Float16*)(ws + off); off += (size_t)NN * HC * sizeof(_Float16);
    _Float16* wp = (_Float16*)(ws + off); off += (size_t)HC * HC * sizeof(_Float16);
    (void)ws_size; (void)n_in; (void)in_sizes; (void)out_size;

    const int T = 256;

    // ---- self-loop edge_attr (mean of incoming) ----
    fill_f32<<<cdiv(NN * (1 + FEDGE), T), T, 0, stream>>>(cnt, 0.0f, NN * (1 + FEDGE));
    edge_count_sum<<<cdiv(EE * FEDGE, T), T, 0, stream>>>(tgt, ea, cnt, asum);
    make_loop_attr<<<cdiv(NN * FEDGE, T), T, 0, stream>>>(cnt, asum, lattr);

    const int gemmBlocks = NN / 80;           // 625 (5 M-tiles per wave)
    const int edgeBlocks = cdiv(EF, T / 32);  // 8 waves per block

    // ================= layer 1 (Fin = 64) =================
    f32_to_f16<<<cdiv(NN * 64, T), T, 0, stream>>>(x, xh, NN * 64);
    pack_w_f16<<<cdiv(64 * HC, T), T, 0, stream>>>(w1l, wp, 64);
    gemm_wmma_f16<64><<<gemmBlocks, T, 0, stream>>>(xh, wp, b1l, xl);
    pack_w_f16<<<cdiv(64 * HC, T), T, 0, stream>>>(w1r, wp, 64);
    gemm_wmma_f16<64><<<gemmBlocks, T, 0, stream>>>(xh, wp, b1r, xr);

    fill_f32<<<cdiv(NN * NHEAD, T), T, 0, stream>>>(mx, -1.0e30f, NN * NHEAD);
    fill_f32<<<cdiv(NN * NHEAD, T), T, 0, stream>>>(den, 0.0f, NN * NHEAD);
    fill_f32<<<cdiv(NN * HC, T), T, 0, stream>>>(agg, 0.0f, NN * HC);

    edge_logit<<<edgeBlocks, T, 0, stream>>>(src, tgt, ea, lattr, xl, xr, w1e, att1, logit, mx);
    edge_aggregate<<<edgeBlocks, T, 0, stream>>>(src, tgt, xl, logit, mx, den, agg);
    node_finalize<<<cdiv(NN * HC, T), T, 0, stream>>>(agg, den, bias1, hbuf);

    // ================= layer 2 (Fin = 128) =================
    f32_to_f16<<<cdiv(NN * HC, T), T, 0, stream>>>(hbuf, xh, NN * HC);
    pack_w_f16<<<cdiv(HC * HC, T), T, 0, stream>>>(w2l, wp, HC);
    gemm_wmma_f16<128><<<gemmBlocks, T, 0, stream>>>(xh, wp, b2l, xl);
    pack_w_f16<<<cdiv(HC * HC, T), T, 0, stream>>>(w2r, wp, HC);
    gemm_wmma_f16<128><<<gemmBlocks, T, 0, stream>>>(xh, wp, b2r, xr);

    fill_f32<<<cdiv(NN * NHEAD, T), T, 0, stream>>>(mx, -1.0e30f, NN * NHEAD);
    fill_f32<<<cdiv(NN * NHEAD, T), T, 0, stream>>>(den, 0.0f, NN * NHEAD);
    fill_f32<<<cdiv(NN * HC, T), T, 0, stream>>>(agg, 0.0f, NN * HC);

    edge_logit<<<edgeBlocks, T, 0, stream>>>(src, tgt, ea, lattr, xl, xr, w2e, att2, logit, mx);
    edge_aggregate<<<edgeBlocks, T, 0, stream>>>(src, tgt, xl, logit, mx, den, agg);
    node_finalize<<<cdiv(NN * HC, T), T, 0, stream>>>(agg, den, bias2, hbuf);

    // ================= history encoder: out = h @ hw + hb =================
    f32_to_f16<<<cdiv(NN * HC, T), T, 0, stream>>>(hbuf, xh, NN * HC);
    pack_w_f16<<<cdiv(HC * HC, T), T, 0, stream>>>(hw, wp, HC);
    gemm_wmma_f16<128><<<gemmBlocks, T, 0, stream>>>(xh, wp, hb, out);
}
